// MultiHeadAttn_7430293422992
// MI455X (gfx1250) — compile-verified, matrix operand order checked by reading, not run on hardware
//
#include <hip/hip_runtime.h>
#include <hip/hip_bf16.h>

// ---------------------------------------------------------------------------
// MI455X (gfx1250) causal MHA block, bf16 WMMA + flash attention.
// T=2048 B=2 D=1024, 16 heads x 64. wave32.
// GEMMs: async direct-to-LDS (ASYNCcnt) double-buffered tiles + WMMA.
// Attention: 32-query tiles per wave, K/V fragments shared by both subtiles.
// ---------------------------------------------------------------------------

#define T_SEQ  2048
#define BATCH  2
#define DMODEL 1024
#define NHEAD  16
#define DHEAD  64
#define ROWS   (T_SEQ * BATCH)          // 4096 token rows (t*B + b)
#define NQKV   (3 * DMODEL)             // fused Q (1024) + KV (2048) columns

typedef __attribute__((ext_vector_type(8)))  float   v8f;
typedef __attribute__((ext_vector_type(8)))  __bf16  bf16x8;
typedef __attribute__((ext_vector_type(16))) __bf16  bf16x16;

static __device__ __forceinline__ __bf16 to_bf16(float f) {
  unsigned u = __builtin_bit_cast(unsigned, f);
  unsigned r = u + 0x7FFFu + ((u >> 16) & 1u);      // round to nearest even
  unsigned short hs = (unsigned short)(r >> 16);
  return __builtin_bit_cast(__bf16, hs);
}

static __device__ __forceinline__ bf16x16 cat8(bf16x8 lo, bf16x8 hi) {
  return __builtin_shufflevector(lo, hi, 0,1,2,3,4,5,6,7,8,9,10,11,12,13,14,15);
}

static __device__ __forceinline__ v8f wmma_bf16(bf16x16 a, bf16x16 b, v8f c) {
  // (neg_a, A, neg_b, B, c_mod, C, reuse_a, reuse_b)
  return __builtin_amdgcn_wmma_f32_16x16x32_bf16(false, a, false, b, (short)0, c,
                                                 false, false);
}

// Async global -> LDS 16B copy (GVS mode: saddr + 32-bit lane offset).
// LDS destination byte address = low 32 bits of the generic pointer
// (flat aperture mapping).
static __device__ __forceinline__ void async_b128(unsigned lds_addr,
                                                  int voff,
                                                  const void* saddr) {
  asm volatile("global_load_async_to_lds_b128 %0, %1, %2"
               :: "v"(lds_addr), "v"(voff), "s"(saddr) : "memory");
}
static __device__ __forceinline__ void wait_async0() {
  asm volatile("s_wait_asynccnt 0" ::: "memory");
}

// --------------------------- prep kernels ----------------------------------

__global__ void k_f32_to_bf16(const float* __restrict__ in,
                              __bf16* __restrict__ out, int n) {
  int i = blockIdx.x * blockDim.x + threadIdx.x;
  if (i < n) out[i] = to_bf16(in[i]);
}

// src f32 [K][N] row-major  ->  dst bf16 [row_ofs + N][K] row-major (N-major)
__global__ void k_transpose_bf16(const float* __restrict__ src,
                                 __bf16* __restrict__ dst,
                                 int K, int N, int row_ofs) {
  __shared__ float tile[32][33];
  const int k0 = blockIdx.x * 32, n0 = blockIdx.y * 32;
  const int tx = threadIdx.x, ty = threadIdx.y;      // (32, 8)
#pragma unroll
  for (int q = 0; q < 4; ++q)
    tile[ty + 8 * q][tx] = src[(size_t)(k0 + ty + 8 * q) * N + (n0 + tx)];
  __syncthreads();
#pragma unroll
  for (int q = 0; q < 4; ++q)
    dst[(size_t)(row_ofs + n0 + ty + 8 * q) * K + (k0 + tx)] =
        to_bf16(tile[tx][ty + 8 * q]);
}

// --------------------------- WMMA GEMM -------------------------------------
// C[M x N] = A[M x K] * Bt[N x K]^T.  Workgroup tile 128x64, 8 waves,
// each wave 32x32 (2x2 of 16x16x32 bf16 WMMA).  Tiles stream into LDS with
// GLOBAL_LOAD_ASYNC_TO_LDS_B128 (ASYNCcnt), ping-pong double-buffered so the
// next K-tile loads while the current one feeds the WMMAs.  LDS row stride
// padded to 40 bf16 (80B -> 20-bank step, conflict-free b128 fragment reads).
// MODE 0: scatter epilogue into Q / K / V^T bf16 buffers (QKV projection).
// MODE 1: plain f32 store (output projection).
template <int MODE>
__global__ __launch_bounds__(256)
void k_gemm_wmma(const __bf16* __restrict__ A, const __bf16* __restrict__ Bt,
                 int Kdim,
                 __bf16* __restrict__ Qb, __bf16* __restrict__ Kb,
                 __bf16* __restrict__ Vt, float* __restrict__ AO) {
  __shared__ __align__(16) __bf16 As[2][128 * 40];
  __shared__ __align__(16) __bf16 Bs[2][64 * 40];

  const int tid  = threadIdx.x;
  const int lane = tid & 31;
  const int wv   = tid >> 5;
  const int lm   = lane & 15;
  const int sel  = lane >> 4;
  const int so   = sel * 8;
  const int row0 = blockIdx.x * 128;
  const int col0 = blockIdx.y * 64;
  const int mo   = (wv >> 1) * 32;
  const int no   = (wv & 1) * 32;

  const int ldrow = tid >> 1, half = tid & 1;
  // per-thread invariant pieces of the async copies
  const int voffA = (ldrow * Kdim + half * 16) * 2;           // bytes
  const int voffB = voffA;                                    // same pattern
  const unsigned ldsA[2] = {
      (unsigned)(size_t)(void*)(&As[0][ldrow * 40 + half * 16]),
      (unsigned)(size_t)(void*)(&As[1][ldrow * 40 + half * 16])};
  const unsigned ldsB[2] = {
      (unsigned)(size_t)(void*)(&Bs[0][ldrow * 40 + half * 16]),
      (unsigned)(size_t)(void*)(&Bs[1][ldrow * 40 + half * 16])};
  const char* baseA = (const char*)(A  + (size_t)row0 * Kdim);
  const char* baseB = (const char*)(Bt + (size_t)col0 * Kdim);

  auto issue_tile = [&](int buf, int kc) {
    const char* sa = baseA + (size_t)kc * 2;
    async_b128(ldsA[buf],      voffA,      sa);
    async_b128(ldsA[buf] + 16, voffA + 16, sa);                // chunk at +16B
    if (tid < 128) {
      const char* sb = baseB + (size_t)kc * 2;
      async_b128(ldsB[buf],      voffB,      sb);
      async_b128(ldsB[buf] + 16, voffB + 16, sb);
    }
  };

  v8f acc[2][2];
#pragma unroll
  for (int i = 0; i < 2; ++i)
#pragma unroll
    for (int j = 0; j < 2; ++j) acc[i][j] = {};

  int buf = 0;
  issue_tile(0, 0);

  for (int kc = 0; kc < Kdim; kc += 32) {
    wait_async0();          // current buffer's async copies (this wave) done
    __syncthreads();        // all waves' copies visible, prior reads retired
    if (kc + 32 < Kdim) issue_tile(buf ^ 1, kc + 32);   // overlap next tile

    bf16x16 afr[2], bfr[2];
#pragma unroll
    for (int i = 0; i < 2; ++i) {
      const __bf16* pa = &As[buf][(mo + i * 16 + lm) * 40 + so];
      afr[i] = cat8(*(const bf16x8*)pa, *(const bf16x8*)(pa + 16));
      const __bf16* pb = &Bs[buf][(no + i * 16 + lm) * 40 + so];
      bfr[i] = cat8(*(const bf16x8*)pb, *(const bf16x8*)(pb + 16));
    }
#pragma unroll
    for (int i = 0; i < 2; ++i)
#pragma unroll
      for (int j = 0; j < 2; ++j)
        acc[i][j] = wmma_bf16(afr[i], bfr[j], acc[i][j]);
    buf ^= 1;
  }

  // epilogue: C layout = vgpr e, lanes 0-15 -> row e, lanes 16-31 -> row 8+e
#pragma unroll
  for (int i = 0; i < 2; ++i) {
#pragma unroll
    for (int j = 0; j < 2; ++j) {
#pragma unroll
      for (int e = 0; e < 8; ++e) {
        const int r = row0 + mo + i * 16 + (sel ? 8 + e : e);
        const int c = col0 + no + j * 16 + lm;
        const float v = acc[i][j][e];
        if (MODE == 0) {
          const int t = r >> 1, b = r & 1;              // r = t*B + b, B=2
          if (c < DMODEL) {                              // Q column
            const int n = c >> 6, d = c & 63;
            Qb[((size_t)(b * NHEAD + n) * T_SEQ + t) * DHEAD + d] = to_bf16(v);
          } else {                                       // KV column
            const int c2 = c - DMODEL;
            const int n = c2 >> 7, wd = c2 & 127;
            const int bn = b * NHEAD + n;
            if (wd < DHEAD)
              Kb[((size_t)bn * T_SEQ + t) * DHEAD + wd] = to_bf16(v);
            else                                         // V stored transposed
              Vt[((size_t)bn * DHEAD + (wd - DHEAD)) * T_SEQ + t] = to_bf16(v);
          }
        } else {
          AO[(size_t)r * DMODEL + c] = v;
        }
      }
    }
  }
}

// --------------------------- flash attention -------------------------------
// 128-thread blocks = 4 independent waves.  Each wave owns one (bn, 32-query)
// tile: two 16-row subtiles that SHARE every K and V fragment (2x the WMMA
// work per byte of K/V traffic from L2).  Private per-wave LDS P-region, no
// block barriers (per-wave s_wait_dscnt only), so divergent causal trip
// counts between waves are safe.
// Per 32-key chunk: S(32x32)=Q*Kt (8 WMMA), online softmax (shfl_xor row
// reductions in the documented C-layout), P restaged via LDS to A-layout,
// O += P*V (8 WMMA against pre-transposed V for contiguous B-fragments).
__global__ __launch_bounds__(128)
void k_attention(const __bf16* __restrict__ Q, const __bf16* __restrict__ K,
                 const __bf16* __restrict__ Vt, __bf16* __restrict__ O) {
  __shared__ __align__(16) __bf16 Pl[4][32 * 40];   // per-wave 32x32 P tile

  const int wv    = threadIdx.x >> 5;      // wave in block
  const int lane  = threadIdx.x & 31;
  const int itile = blockIdx.x * 4 + wv;   // 0..63 (32-query tiles)
  const int bn    = blockIdx.y;            // 0..31 = b*16 + n
  const int b     = bn >> 4, n = bn & 15;
  const int lm    = lane & 15;
  const int sel   = lane >> 4;
  const int so    = sel * 8;
  const int i0    = itile * 32;

  const __bf16* Qp = Q  + (size_t)bn * T_SEQ * DHEAD;
  const __bf16* Kp = K  + (size_t)bn * T_SEQ * DHEAD;
  const __bf16* Vp = Vt + (size_t)bn * DHEAD * T_SEQ;

  // Q fragments: 2 subtiles x 2 d-chunks (A operand)
  bf16x16 qf[2][2];
#pragma unroll
  for (int mi = 0; mi < 2; ++mi) {
    const __bf16* qr = Qp + (size_t)(i0 + mi * 16 + lm) * DHEAD;
#pragma unroll
    for (int kk = 0; kk < 2; ++kk) {
      const __bf16* p = qr + kk * 32 + so;
      qf[mi][kk] = cat8(*(const bf16x8*)p, *(const bf16x8*)(p + 16));
    }
  }

  float mrow[2][8], lrow[2][8];
  v8f accO[2][4];
#pragma unroll
  for (int mi = 0; mi < 2; ++mi) {
#pragma unroll
    for (int i = 0; i < 8; ++i) { mrow[mi][i] = -1e30f; lrow[mi][i] = 0.f; }
#pragma unroll
    for (int d = 0; d < 4; ++d) accO[mi][d] = {};
  }

  const int jend = i0 + 31;                         // causal upper bound
  for (int jc = 0; jc <= jend; jc += 32) {
    v8f s[2][2] = {{{}, {}}, {{}, {}}};
#pragma unroll
    for (int kk = 0; kk < 2; ++kk) {                // K frags shared by both mi
      const __bf16* k0p = Kp + (size_t)(jc + lm) * DHEAD + kk * 32 + so;
      const __bf16* k1p = Kp + (size_t)(jc + 16 + lm) * DHEAD + kk * 32 + so;
      bf16x16 kf0 = cat8(*(const bf16x8*)k0p, *(const bf16x8*)(k0p + 16));
      bf16x16 kf1 = cat8(*(const bf16x8*)k1p, *(const bf16x8*)(k1p + 16));
#pragma unroll
      for (int mi = 0; mi < 2; ++mi) {
        s[mi][0] = wmma_bf16(qf[mi][kk], kf0, s[mi][0]);
        s[mi][1] = wmma_bf16(qf[mi][kk], kf1, s[mi][1]);
      }
    }

#pragma unroll
    for (int mi = 0; mi < 2; ++mi) {
      // scale + causal mask + per-row chunk max
      float rmax[8];
#pragma unroll
      for (int i = 0; i < 8; ++i) {
        const int m  = mi * 16 + (sel ? 8 + i : i);
        const int ig = i0 + m;
        float a  = s[mi][0][i] * 0.125f;             // 1/sqrt(64)
        float bb = s[mi][1][i] * 0.125f;
        if (jc + lm > ig)      a  = -1e30f;
        if (jc + 16 + lm > ig) bb = -1e30f;
        s[mi][0][i] = a; s[mi][1][i] = bb;
        rmax[i] = fmaxf(a, bb);
      }
#pragma unroll
      for (int off = 1; off < 16; off <<= 1)
#pragma unroll
        for (int i = 0; i < 8; ++i)
          rmax[i] = fmaxf(rmax[i], __shfl_xor(rmax[i], off));

      float psum[8];
#pragma unroll
      for (int i = 0; i < 8; ++i) {
        const float mn   = fmaxf(mrow[mi][i], rmax[i]);
        const float corr = __expf(mrow[mi][i] - mn);
        const float p0   = __expf(s[mi][0][i] - mn);
        const float p1   = __expf(s[mi][1][i] - mn);
        mrow[mi][i] = mn;
        psum[i] = p0 + p1;
        lrow[mi][i] *= corr;
#pragma unroll
        for (int d = 0; d < 4; ++d) accO[mi][d][i] *= corr;
        const int m = mi * 16 + (sel ? 8 + i : i);   // C-layout -> LDS [32][40]
        Pl[wv][m * 40 + lm]      = to_bf16(p0);
        Pl[wv][m * 40 + 16 + lm] = to_bf16(p1);
      }
#pragma unroll
      for (int off = 1; off < 16; off <<= 1)
#pragma unroll
        for (int i = 0; i < 8; ++i)
          psum[i] += __shfl_xor(psum[i], off);
#pragma unroll
      for (int i = 0; i < 8; ++i) lrow[mi][i] += psum[i];
    }

    asm volatile("s_wait_dscnt 0" ::: "memory");     // LDS writes visible

    bf16x16 pf[2];
#pragma unroll
    for (int mi = 0; mi < 2; ++mi) {                 // P as A-fragments
      const __bf16* pp = &Pl[wv][(mi * 16 + lm) * 40 + so];
      pf[mi] = cat8(*(const bf16x8*)pp, *(const bf16x8*)(pp + 16));
    }
#pragma unroll
    for (int d = 0; d < 4; ++d) {                    // V frags shared by both mi
      const __bf16* vp = Vp + (size_t)(d * 16 + lm) * T_SEQ + jc + so;
      bf16x16 vf = cat8(*(const bf16x8*)vp, *(const bf16x8*)(vp + 16));
#pragma unroll
      for (int mi = 0; mi < 2; ++mi)
        accO[mi][d] = wmma_bf16(pf[mi], vf, accO[mi][d]);
    }
  }

  // normalize + store attn_vec rows (t*B + b, col = n*64 + d)
#pragma unroll
  for (int mi = 0; mi < 2; ++mi) {
#pragma unroll
    for (int i = 0; i < 8; ++i) {
      const int m   = mi * 16 + (sel ? 8 + i : i);
      const float inv = 1.0f / lrow[mi][i];
      const int r   = (i0 + m) * BATCH + b;
#pragma unroll
      for (int d = 0; d < 4; ++d) {
        const int c = n * DHEAD + d * 16 + lm;
        O[(size_t)r * DMODEL + c] = to_bf16(accO[mi][d][i] * inv);
      }
    }
  }
}

// --------------------------- residual + LayerNorm --------------------------

__global__ __launch_bounds__(256)
void k_layernorm(const float* __restrict__ h, const float* __restrict__ ao,
                 const float* __restrict__ g, const float* __restrict__ be,
                 float* __restrict__ out) {
  __shared__ float red[256];
  const int r = blockIdx.x, tid = threadIdx.x;
  const float* hr = h  + (size_t)r * DMODEL;
  const float* ar = ao + (size_t)r * DMODEL;
  float x[4]; float s = 0.f;
#pragma unroll
  for (int q = 0; q < 4; ++q) { x[q] = hr[tid + q * 256] + ar[tid + q * 256]; s += x[q]; }
  red[tid] = s; __syncthreads();
  for (int st = 128; st > 0; st >>= 1) { if (tid < st) red[tid] += red[tid + st]; __syncthreads(); }
  const float mu = red[0] * (1.0f / DMODEL);
  __syncthreads();
  float v = 0.f;
#pragma unroll
  for (int q = 0; q < 4; ++q) { const float d = x[q] - mu; v += d * d; }
  red[tid] = v; __syncthreads();
  for (int st = 128; st > 0; st >>= 1) { if (tid < st) red[tid] += red[tid + st]; __syncthreads(); }
  const float rstd = rsqrtf(red[0] * (1.0f / DMODEL) + 1e-5f);
#pragma unroll
  for (int q = 0; q < 4; ++q) {
    const int c = tid + q * 256;
    out[(size_t)r * DMODEL + c] = (x[q] - mu) * rstd * g[c] + be[c];
  }
}

// --------------------------- launch ----------------------------------------

extern "C" void kernel_launch(void* const* d_in, const int* in_sizes, int n_in,
                              void* d_out, int out_size, void* d_ws, size_t ws_size,
                              hipStream_t stream) {
  (void)in_sizes; (void)n_in; (void)out_size; (void)ws_size;
  const float* h    = (const float*)d_in[0];
  // d_in[1] = attn_mask (plain causal) -- recomputed analytically, not read
  const float* Wq   = (const float*)d_in[2];
  const float* Wkv  = (const float*)d_in[3];
  const float* Wo   = (const float*)d_in[4];
  const float* ln_g = (const float*)d_in[5];
  const float* ln_b = (const float*)d_in[6];
  float* out = (float*)d_out;

  char* wp = (char*)d_ws;
  auto carve = [&](size_t bytes) {
    char* p = wp; wp += (bytes + 255) & ~(size_t)255; return p;
  };
  __bf16* Xbf   = (__bf16*)carve((size_t)ROWS * DMODEL * 2);            // h in bf16
  __bf16* WtQKV = (__bf16*)carve((size_t)NQKV * DMODEL * 2);            // [Wq;Wkv]^T
  __bf16* WtO   = (__bf16*)carve((size_t)DMODEL * DMODEL * 2);          // Wo^T
  __bf16* Qb    = (__bf16*)carve((size_t)BATCH * NHEAD * T_SEQ * DHEAD * 2);
  __bf16* Kb    = (__bf16*)carve((size_t)BATCH * NHEAD * T_SEQ * DHEAD * 2);
  __bf16* Vt    = (__bf16*)carve((size_t)BATCH * NHEAD * DHEAD * T_SEQ * 2);
  __bf16* Obuf  = (__bf16*)carve((size_t)ROWS * DMODEL * 2);            // attn_vec
  float*  AO    = (float*)carve((size_t)ROWS * DMODEL * 4);             // attn_out

  // 1) precision staging
  k_f32_to_bf16<<<(ROWS * DMODEL) / 256, 256, 0, stream>>>(h, Xbf, ROWS * DMODEL);
  dim3 tb(32, 8);
  k_transpose_bf16<<<dim3(DMODEL / 32, DMODEL / 32), tb, 0, stream>>>(
      Wq, WtQKV, DMODEL, DMODEL, 0);
  k_transpose_bf16<<<dim3(DMODEL / 32, (2 * DMODEL) / 32), tb, 0, stream>>>(
      Wkv, WtQKV, DMODEL, 2 * DMODEL, DMODEL);
  k_transpose_bf16<<<dim3(DMODEL / 32, DMODEL / 32), tb, 0, stream>>>(
      Wo, WtO, DMODEL, DMODEL, 0);

  // 2) fused QKV projection (WMMA + async LDS) with Q/K/V^T scatter epilogue
  k_gemm_wmma<0><<<dim3(ROWS / 128, NQKV / 64), 256, 0, stream>>>(
      Xbf, WtQKV, DMODEL, Qb, Kb, Vt, nullptr);

  // 3) flash attention (WMMA), 4 waves per block, wave per (bn, 32-query tile)
  k_attention<<<dim3(T_SEQ / 32 / 4, BATCH * NHEAD), 128, 0, stream>>>(
      Qb, Kb, Vt, Obuf);

  // 4) output projection (WMMA + async LDS)
  k_gemm_wmma<1><<<dim3(ROWS / 128, DMODEL / 64), 256, 0, stream>>>(
      Obuf, WtO, DMODEL, nullptr, nullptr, nullptr, AO);

  // 5) residual + LayerNorm
  k_layernorm<<<ROWS, 256, 0, stream>>>(h, AO, ln_g, ln_b, out);
}
